// wholeModel_58162447123364
// MI455X (gfx1250) — matrix operand order repaired and checked by reading, not compiled, not run on hardware
//
#include <hip/hip_runtime.h>
#include <hip/hip_bf16.h>
#include <math.h>

typedef __attribute__((ext_vector_type(16))) _Float16 v16h;
typedef __attribute__((ext_vector_type(8)))  float    v8f;

union Frag16 { v16h v; unsigned u[8]; };
union PackH2 { _Float16 h[2]; unsigned u; };

#define BB   64
#define TT   32
#define DD   4096
#define CC   50
#define AA   300
#define HH   256
#define G3   768     // 3*H
#define FF   512     // 2*H
#define SEGN 32
#define ROWS (BB*SEGN)        // 2048
#define M2   (BB*CC*SEGN)     // 102400
#define BC   (BB*CC)          // 3200

// ---------------------------------------------------------------------------
// 1) segment selection: stable compaction by mean-threshold, emit f16 rows
// ---------------------------------------------------------------------------
__global__ void k_select(const float* __restrict__ att,
                         const float* __restrict__ feat,
                         _Float16* __restrict__ valid16) {
  __shared__ int map[SEGN];
  const int b = blockIdx.x;
  if (threadIdx.x == 0) {
    float s = 0.f;
    for (int t = 0; t < TT; ++t) s += att[b * TT + t];
    const float thr = s / (float)TT;
    int pos = 0;
    for (int t = 0; t < TT; ++t)
      if (att[b * TT + t] >= thr) map[pos++] = t;
    for (; pos < SEGN; ++pos) map[pos] = -1;
  }
  __syncthreads();
  const size_t base = (size_t)b * SEGN * DD;
  for (int i = threadIdx.x; i < SEGN * DD; i += blockDim.x) {
    const int slot = i >> 12, d = i & (DD - 1);
    const int src = map[slot];
    const float v = (src >= 0) ? feat[base + (size_t)src * DD + d] : 0.f;
    valid16[base + i] = (_Float16)v;
  }
}

// ---------------------------------------------------------------------------
// 2) f32 -> f16 convert with optional zero padding (R,C) -> (Rp,Cp)
// ---------------------------------------------------------------------------
__global__ void k_cvt_pad(const float* __restrict__ src, _Float16* __restrict__ dst,
                          int R, int Cc, int Rp, int Cp) {
  const int i = blockIdx.x * blockDim.x + threadIdx.x;
  if (i >= Rp * Cp) return;
  const int r = i / Cp, c = i % Cp;
  const float v = (r < R && c < Cc) ? src[r * Cc + c] : 0.f;
  dst[i] = (_Float16)v;
}

// ---------------------------------------------------------------------------
// 3) generic WMMA GEMM:  D[M,N] = A[M,K] * Bw[N,K]^T (+bias) (+relu)
//    one wave per 16x64 strip (4 accumulators, A fragment reused 4x).
//    K multiple of 32; M multiple of 16; N multiple of 64.
//    store_sb: row m=(b*32+s) is scattered to D[(s*64+b)*ldd + n] (gi layout)
// ---------------------------------------------------------------------------
__global__ __launch_bounds__(32)
void k_gemm(const _Float16* __restrict__ A, int lda,
            const _Float16* __restrict__ Bw, int ldb,
            const float* __restrict__ bias,
            float* __restrict__ D, int ldd,
            int K, int relu, int store_sb) {
  const int m0 = blockIdx.x * 16, n0 = blockIdx.y * 64;
  const int lane = threadIdx.x;
  const int half = lane >> 4, l15 = lane & 15;
  const int kgrpA = half * 8, kgrpB = half * 16;
  const _Float16* aRow = A + (size_t)(m0 + l15) * lda;
  const _Float16* bRow[4];
#pragma unroll
  for (int t = 0; t < 4; ++t)
    bRow[t] = Bw + (size_t)(n0 + t * 16 + l15) * ldb;

  v8f accs[4] = {{}, {}, {}, {}};
  for (int kb = 0; kb < K; kb += 32) {
    Frag16 a;
#pragma unroll
    for (int v = 0; v < 8; ++v) {
      const int ka = kb + kgrpA + ((v & 4) ? 16 : 0) + ((v & 3) << 1);
      a.u[v] = *(const unsigned*)(aRow + ka);
    }
    // prefetch next K-step of B (speculative; tail overshoot is dropped)
    __builtin_prefetch(bRow[0] + kb + 32 + kgrpB, 0, 0);
#pragma unroll
    for (int t = 0; t < 4; ++t) {
      Frag16 bm;
#pragma unroll
      for (int v = 0; v < 8; ++v)
        bm.u[v] = *(const unsigned*)(bRow[t] + kb + kgrpA + ((v & 4) ? 16 : 0) + ((v & 3) << 1));
      accs[t] = __builtin_amdgcn_wmma_f32_16x16x32_f16(false, a.v, false, bm.v,
                                                       (short)0, accs[t], false, false);
    }
  }

#pragma unroll
  for (int t = 0; t < 4; ++t) {
    const int nc = n0 + t * 16 + l15;
    const float bi = bias ? bias[nc] : 0.f;
#pragma unroll
    for (int v = 0; v < 8; ++v) {
      const int m = m0 + v + half * 8;
      float val = accs[t][v] + bi;
      if (relu) val = fmaxf(val, 0.f);
      if (store_sb) {
        const int bI = m >> 5, sI = m & 31;
        D[((size_t)sI * BB + bI) * ldd + nc] = val;
      } else {
        D[(size_t)m * ldd + nc] = val;
      }
    }
  }
}

// ---------------------------------------------------------------------------
// 4) bidirectional GRU recurrence; one 16-wave workgroup per direction.
//    h kept in LDS (f32 + f16 mirror); per-step gh = h*Whh^T via WMMA,
//    each wave handling 16x64 strips (48 units over 16 waves).
// ---------------------------------------------------------------------------
#define SMEM_H32  (BB * HH * 4)          // 65536
#define SMEM_GH   (BB * G3 * 4)          // 196608
#define SMEM_H16  (BB * HH * 2)          // 32768
#define SMEM_GRU  (SMEM_H32 + SMEM_GH + SMEM_H16)   // 294912 <= 320KB

__global__ __launch_bounds__(512)
void k_gru(const float* __restrict__ giF, const float* __restrict__ giB,
           const _Float16* __restrict__ whhF16, const _Float16* __restrict__ whhB16,
           const float* __restrict__ bhhF, const float* __restrict__ bhhB,
           float* __restrict__ feats) {
  extern __shared__ char smem[];
  float*    h32 = (float*)smem;
  float*    gh  = (float*)(smem + SMEM_H32);
  _Float16* h16 = (_Float16*)(smem + SMEM_H32 + SMEM_GH);

  const int dir = blockIdx.x;
  const float*    gi    = dir ? giB : giF;
  const _Float16* whh16 = dir ? whhB16 : whhF16;
  const float*    bhh   = dir ? bhhB : bhhF;

  for (int e = threadIdx.x; e < BB * HH; e += blockDim.x) {
    h32[e] = 0.f; h16[e] = (_Float16)0.f;
  }
  __syncthreads();

  const int lane = threadIdx.x & 31;
  const int wave = threadIdx.x >> 5;
  const int half = lane >> 4, l15 = lane & 15;
  const int kgrpA = half * 8, kgrpB = half * 16;

  for (int step = 0; step < SEGN; ++step) {
    const int sI = dir ? (SEGN - 1 - step) : step;

    // gh = h * Whh^T : [64,256]x[256,768] -> 4 m-tiles x 12 n-strips(64)
    for (int unit = wave; unit < 48; unit += 16) {
      const int m0 = (unit & 3) * 16;
      const int n0 = (unit >> 2) * 64;
      const _Float16* aRow = h16 + (size_t)(m0 + l15) * HH;
      v8f accs[4] = {{}, {}, {}, {}};
#pragma unroll
      for (int kb = 0; kb < HH; kb += 32) {
        Frag16 a;
#pragma unroll
        for (int v = 0; v < 8; ++v) {
          const int ka = kb + kgrpA + ((v & 4) ? 16 : 0) + ((v & 3) << 1);
          a.u[v] = *(const unsigned*)(aRow + ka);
        }
#pragma unroll
        for (int t = 0; t < 4; ++t) {
          const _Float16* bRow = whh16 + (size_t)(n0 + t * 16 + l15) * HH;
          Frag16 bm;
#pragma unroll
          for (int v = 0; v < 8; ++v)
            bm.u[v] = *(const unsigned*)(bRow + kb + kgrpA + ((v & 4) ? 16 : 0) + ((v & 3) << 1));
          accs[t] = __builtin_amdgcn_wmma_f32_16x16x32_f16(false, a.v, false, bm.v,
                                                           (short)0, accs[t], false, false);
        }
      }
#pragma unroll
      for (int t = 0; t < 4; ++t)
#pragma unroll
        for (int v = 0; v < 8; ++v)
          gh[(m0 + v + half * 8) * G3 + n0 + t * 16 + l15] = accs[t][v];
    }
    __syncthreads();

    // gate math (torch gate order r,z,n); gi already includes bih
    for (int e = threadIdx.x; e < BB * HH; e += blockDim.x) {
      const int row = e >> 8, j = e & (HH - 1);
      const float* giRow = gi + ((size_t)sI * BB + row) * G3;
      const float gr = giRow[j]          + gh[row * G3 + j]          + bhh[j];
      const float gz = giRow[HH + j]     + gh[row * G3 + HH + j]     + bhh[HH + j];
      const float gnh =                    gh[row * G3 + 2*HH + j]   + bhh[2*HH + j];
      const float r = 1.f / (1.f + expf(-gr));
      const float z = 1.f / (1.f + expf(-gz));
      const float n = tanhf(giRow[2*HH + j] + r * gnh);
      const float hnew = (1.f - z) * n + z * h32[e];
      h32[e] = hnew;
      h16[e] = (_Float16)hnew;
      feats[((size_t)row * SEGN + sI) * FF + dir * HH + j] = hnew;
    }
    __syncthreads();
  }
}

// ---------------------------------------------------------------------------
// 5) s_score init to fc4 bias
// ---------------------------------------------------------------------------
__global__ void k_init_s(float* __restrict__ s, const float* __restrict__ fc4_b) {
  const int i = blockIdx.x * blockDim.x + threadIdx.x;
  if (i < M2) s[i] = fc4_b[0];
}

// ---------------------------------------------------------------------------
// 6) fused relation head: A = (feats-attr)^2 built on the fly (f16, built
//    once per K-step and reused across a 16x64 strip of fc3 columns),
//    h = relu(A*fc3^T + b3) via WMMA, dot with fc4_w, cross-lane reduce,
//    accumulate into s_score[row] with atomics.
// ---------------------------------------------------------------------------
__global__ __launch_bounds__(32)
void k_fc3fc4(const float* __restrict__ feats, const float* __restrict__ attr,
              const _Float16* __restrict__ fc3w16,
              const float* __restrict__ fc3_b,
              const float* __restrict__ fc4_w,
              float* __restrict__ sscore) {
  const int m0 = blockIdx.x * 16;          // rows over 102400
  const int n0 = blockIdx.y * 64;          // cols over 256
  const int lane = threadIdx.x;
  const int half = lane >> 4, l15 = lane & 15;
  const int kgrpA = half * 8, kgrpB = half * 16;

  const int row = m0 + l15;
  const int bc = row >> 5, sI = row & 31;
  const int bI = bc / CC, cI = bc % CC;
  const float* fRow = feats + ((size_t)bI * SEGN + sI) * FF;
  const float* aRow = attr + (size_t)cI * FF;

  v8f accs[4] = {{}, {}, {}, {}};
  for (int kb = 0; kb < FF; kb += 32) {
    Frag16 a;
#pragma unroll
    for (int v = 0; v < 8; ++v) {
      const int k = kb + kgrpA + ((v & 4) ? 16 : 0) + ((v & 3) << 1);
      const float d0 = fRow[k]     - aRow[k];
      const float d1 = fRow[k + 1] - aRow[k + 1];
      PackH2 p;
      p.h[0] = (_Float16)(d0 * d0);
      p.h[1] = (_Float16)(d1 * d1);
      a.u[v] = p.u;
    }
#pragma unroll
    for (int t = 0; t < 4; ++t) {
      const _Float16* bRow = fc3w16 + (size_t)(n0 + t * 16 + l15) * FF;
      Frag16 bm;
#pragma unroll
      for (int v = 0; v < 8; ++v)
        bm.u[v] = *(const unsigned*)(bRow + kb + kgrpA + ((v & 4) ? 16 : 0) + ((v & 3) << 1));
      accs[t] = __builtin_amdgcn_wmma_f32_16x16x32_f16(false, a.v, false, bm.v,
                                                       (short)0, accs[t], false, false);
    }
  }

  float con[8] = {0.f, 0.f, 0.f, 0.f, 0.f, 0.f, 0.f, 0.f};
#pragma unroll
  for (int t = 0; t < 4; ++t) {
    const int nc = n0 + t * 16 + l15;
    const float b3 = fc3_b[nc];
    const float w4 = fc4_w[nc];
#pragma unroll
    for (int v = 0; v < 8; ++v)
      con[v] += fmaxf(accs[t][v] + b3, 0.f) * w4;
  }
#pragma unroll
  for (int v = 0; v < 8; ++v)
    for (int o = 1; o < 16; o <<= 1)
      con[v] += __shfl_xor(con[v], o, 32);
  if (l15 == 0) {
#pragma unroll
    for (int v = 0; v < 8; ++v)
      atomicAdd(&sscore[m0 + v + half * 8], con[v]);
  }
}

// ---------------------------------------------------------------------------
// 7) softmax over axis-0 (size 3200) per segment column, then mean over cols
// ---------------------------------------------------------------------------
__global__ void k_colsoft(const float* __restrict__ s,
                          float* __restrict__ colmax, float* __restrict__ colsum) {
  __shared__ float red[256];
  const int j = blockIdx.x;
  float m = -1e30f;
  for (int i = threadIdx.x; i < BC; i += 256) m = fmaxf(m, s[(size_t)i * SEGN + j]);
  red[threadIdx.x] = m; __syncthreads();
  for (int o = 128; o > 0; o >>= 1) {
    if (threadIdx.x < o) red[threadIdx.x] = fmaxf(red[threadIdx.x], red[threadIdx.x + o]);
    __syncthreads();
  }
  m = red[0]; __syncthreads();
  float z = 0.f;
  for (int i = threadIdx.x; i < BC; i += 256) z += expf(s[(size_t)i * SEGN + j] - m);
  red[threadIdx.x] = z; __syncthreads();
  for (int o = 128; o > 0; o >>= 1) {
    if (threadIdx.x < o) red[threadIdx.x] += red[threadIdx.x + o];
    __syncthreads();
  }
  if (threadIdx.x == 0) { colmax[j] = m; colsum[j] = red[0]; }
}

__global__ void k_out(const float* __restrict__ s,
                      const float* __restrict__ colmax, const float* __restrict__ colsum,
                      float* __restrict__ out) {
  const int i = blockIdx.x * blockDim.x + threadIdx.x;
  if (i >= BC) return;
  float acc = 0.f;
  for (int j = 0; j < SEGN; ++j)
    acc += expf(s[(size_t)i * SEGN + j] - colmax[j]) / colsum[j];
  out[i] = acc / (float)SEGN;
}

// ---------------------------------------------------------------------------
extern "C" void kernel_launch(void* const* d_in, const int* in_sizes, int n_in,
                              void* d_out, int out_size, void* d_ws, size_t ws_size,
                              hipStream_t stream) {
  const float* feat   = (const float*)d_in[0];
  const float* att    = (const float*)d_in[1];
  const float* attrib = (const float*)d_in[2];
  const float* fc1w = (const float*)d_in[3];  const float* fc1b = (const float*)d_in[4];
  const float* fc2w = (const float*)d_in[5];  const float* fc2b = (const float*)d_in[6];
  const float* fc3w = (const float*)d_in[7];  const float* fc3b = (const float*)d_in[8];
  const float* fc4w = (const float*)d_in[9];  const float* fc4b = (const float*)d_in[10];
  const float* wihF = (const float*)d_in[11]; const float* whhF = (const float*)d_in[12];
  const float* bihF = (const float*)d_in[13]; const float* bhhF = (const float*)d_in[14];
  const float* wihB = (const float*)d_in[15]; const float* whhB = (const float*)d_in[16];
  const float* bihB = (const float*)d_in[17]; const float* bhhB = (const float*)d_in[18];
  float* out = (float*)d_out;

  char* ws = (char*)d_ws;
  size_t off = 0;
  auto carve = [&](size_t bytes) -> void* {
    off = (off + 255) & ~(size_t)255;
    void* p = ws + off;
    off += bytes;
    return p;
  };

  _Float16* valid16  = (_Float16*)carve((size_t)ROWS * DD * 2);
  _Float16* wihF16   = (_Float16*)carve((size_t)G3 * DD * 2);
  _Float16* wihB16   = (_Float16*)carve((size_t)G3 * DD * 2);
  _Float16* whhF16   = (_Float16*)carve((size_t)G3 * HH * 2);
  _Float16* whhB16   = (_Float16*)carve((size_t)G3 * HH * 2);
  float*    giF      = (float*)carve((size_t)SEGN * BB * G3 * 4);
  float*    giB      = (float*)carve((size_t)SEGN * BB * G3 * 4);
  float*    featsV   = (float*)carve((size_t)BB * SEGN * FF * 4);
  _Float16* attrP16  = (_Float16*)carve((size_t)64 * 320 * 2);
  _Float16* fc1wP16  = (_Float16*)carve((size_t)4096 * 320 * 2);
  float*    attr1    = (float*)carve((size_t)64 * 4096 * 4);
  _Float16* attr1_16 = (_Float16*)carve((size_t)64 * 4096 * 2);
  _Float16* fc2w16   = (_Float16*)carve((size_t)512 * 4096 * 2);
  float*    attr2    = (float*)carve((size_t)64 * 512 * 4);
  _Float16* fc3w16   = (_Float16*)carve((size_t)256 * 512 * 2);
  float*    sscore   = (float*)carve((size_t)M2 * 4);
  float*    colmax   = (float*)carve(32 * 4);
  float*    colsum   = (float*)carve(32 * 4);
  (void)ws_size; (void)n_in; (void)in_sizes; (void)out_size;

  // stage f16 operands
  k_select<<<BB, 256, 0, stream>>>(att, feat, valid16);
  auto cvt = [&](const float* s, _Float16* d, int R, int C, int Rp, int Cp) {
    const int tot = Rp * Cp;
    k_cvt_pad<<<(tot + 255) / 256, 256, 0, stream>>>(s, d, R, C, Rp, Cp);
  };
  cvt(wihF, wihF16, G3, DD, G3, DD);
  cvt(wihB, wihB16, G3, DD, G3, DD);
  cvt(whhF, whhF16, G3, HH, G3, HH);
  cvt(whhB, whhB16, G3, HH, G3, HH);
  cvt(fc1w, fc1wP16, 4096, AA, 4096, 320);
  cvt(attrib, attrP16, CC, AA, 64, 320);
  cvt(fc2w, fc2w16, 512, 4096, 512, 4096);
  cvt(fc3w, fc3w16, 256, 512, 256, 512);

  // gi = x * Wih^T + bih (both directions), scattered to [s][b][g]
  {
    dim3 g(ROWS / 16, G3 / 64);
    k_gemm<<<g, 32, 0, stream>>>(valid16, DD, wihF16, DD, bihF, giF, G3, DD, 0, 1);
    k_gemm<<<g, 32, 0, stream>>>(valid16, DD, wihB16, DD, bihB, giB, G3, DD, 0, 1);
  }

  // semantic net: attr = relu(relu(attrib*fc1^T+b1)*fc2^T+b2)
  {
    dim3 g1(64 / 16, 4096 / 64);
    k_gemm<<<g1, 32, 0, stream>>>(attrP16, 320, fc1wP16, 320, fc1b, attr1, 4096, 320, 1, 0);
    cvt(attr1, attr1_16, 64, 4096, 64, 4096);
    dim3 g2(64 / 16, 512 / 64);
    k_gemm<<<g2, 32, 0, stream>>>(attr1_16, 4096, fc2w16, 4096, fc2b, attr2, 512, 4096, 1, 0);
  }

  // bidirectional GRU recurrence -> feats [64,32,512]
  k_gru<<<2, 512, SMEM_GRU, stream>>>(giF, giB, whhF16, whhB16, bhhF, bhhB, featsV);

  // relation head (fused fc3+fc4)
  k_init_s<<<(M2 + 255) / 256, 256, 0, stream>>>(sscore, fc4b);
  {
    dim3 g(M2 / 16, 256 / 64);
    k_fc3fc4<<<g, 32, 0, stream>>>(featsV, attr2, fc3w16, fc3b, fc4w, sscore);
  }

  // softmax over dim 0 per segment column, then mean
  k_colsoft<<<SEGN, 256, 0, stream>>>(sscore, colmax, colsum);
  k_out<<<(BC + 255) / 256, 256, 0, stream>>>(sscore, colmax, colsum, out);
}
